// RNNClass_31018253811769
// MI455X (gfx1250) — matrix operand order repaired and checked by reading, not compiled, not run on hardware
//
#include <hip/hip_runtime.h>
#include <cstdint>
#include <cmath>

#define B_   128
#define T_   512
#define HS_  512
#define G4_  2048
#define NC_  5
#define KTOT (T_*HS_)      // 262144
#define KC_  64
#define KCH  (KTOT/KC_)    // 4096 per K-chunk

typedef __attribute__((ext_vector_type(16))) __bf16 v16bf;
typedef __attribute__((ext_vector_type(8)))  float  v8f;

union ABu { uint4 u[2]; v16bf v; __bf16 h[16]; };

__device__ inline __bf16 f2bf(float f) {
  unsigned u = __builtin_bit_cast(unsigned, f);
  u += 0x7fffu + ((u >> 16) & 1u);          // round-to-nearest-even
  unsigned short s = (unsigned short)(u >> 16);
  return __builtin_bit_cast(__bf16, s);
}
__device__ inline v8f zero8() { v8f z = {0.f,0.f,0.f,0.f,0.f,0.f,0.f,0.f}; return z; }

// branch-free gate math: v_exp_f32 + v_rcp_f32, no OCML slow paths
__device__ inline float sigm_f(float x) {
  return __builtin_amdgcn_rcpf(1.0f + __expf(-x));
}
__device__ inline float tanh_f(float x) {
  return 1.0f - 2.0f * __builtin_amdgcn_rcpf(1.0f + __expf(2.0f * x));
}

// ---------------------------------------------------------------------------
// Prep: bias = b_ih + b_hh; swizzle w_hh (2048x512 f32) into WMMA-B bf16 tiles.
// Tile (nt,kt): lane<->N (nt*16+lane%16), slot i<->K (table order); 1KB per tile.
// ---------------------------------------------------------------------------
__global__ void prep_kernel(const float* __restrict__ w_hh,
                            const float* __restrict__ b_ih,
                            const float* __restrict__ b_hh,
                            __bf16* __restrict__ ws_whh,
                            float* __restrict__ ws_bias) {
  int gid = blockIdx.x * blockDim.x + threadIdx.x;
  if (gid < G4_) ws_bias[gid] = b_ih[gid] + b_hh[gid];
  if (gid < G4_ * HS_) {
    int tile = gid >> 9;        // nt*16 + kt
    int rem  = gid & 511;
    int lane = rem >> 4;
    int i    = rem & 15;
    int kt   = tile & 15;
    int nt   = tile >> 4;
    int n    = nt * 16 + (lane & 15);
    int k    = kt * 32 + (lane >> 4) * 8 + i + (i & 8);
    ws_whh[gid] = f2bf(w_hh[n * HS_ + k]);
  }
}

// ---------------------------------------------------------------------------
// LSTM: 8 WGs x 16 batch rows, 1024 threads (32 waves). Wave w owns hidden
// cols [w*16, w*16+16) for ALL four gates -> acc[4] tiles, c-state in VGPRs.
// Per step: gates(16x2048) = h(16x512 bf16, LDS double buffer) @ whhB via
// v_wmma_f32_16x16x32_bf16. An opaque zero OFFSET (not the base pointer) is
// laundered per step: defeats LICM hoist+spill of the t-invariant weight
// loads while preserving global address-space provenance -> global_load_b128.
// ---------------------------------------------------------------------------
__global__ void __launch_bounds__(1024, 1)
lstm_kernel(const float* __restrict__ x,
            const float* __restrict__ w_ih,
            const __bf16* __restrict__ whhB,
            const float* __restrict__ bias,
            __bf16* __restrict__ flat) {
  __shared__ __bf16 hbuf[2][16 * HS_];     // 32 KB double buffer
  const int tid   = threadIdx.x;
  const int lane  = tid & 31;
  const int w     = tid >> 5;              // wave 0..31 -> hidden cols [w*16, w*16+16)
  const int l16   = lane & 15;
  const int hi    = lane >> 4;
  const int bbase = blockIdx.x * 16;

  for (int i = tid; i < 16 * HS_; i += blockDim.x) hbuf[0][i] = f2bf(0.f);
  __syncthreads();

  // per-column input-projection constants (invariant over t)
  float wih_r[4], bias_r[4];
#pragma unroll
  for (int g = 0; g < 4; ++g) {
    int n = g * HS_ + w * 16 + l16;
    wih_r[g]  = w_ih[n];
    bias_r[g] = bias[n];
  }

  v8f c_st = zero8();
  v8f acc[4];
  size_t zoff = 0;                         // always 0, but opaque to the compiler

#pragma unroll 1
  for (int t = 0; t < T_; ++t) {
    asm volatile("" : "+s"(zoff));         // re-launder each step
    const __bf16* wb = whhB + zoff;        // provenance kept -> global loads

    const __bf16* hin  = hbuf[t & 1];
    __bf16*       hout = hbuf[(t + 1) & 1];
#pragma unroll
    for (int g = 0; g < 4; ++g) acc[g] = zero8();

#pragma unroll 2
    for (int kt = 0; kt < 16; ++kt) {
      ABu a;                                         // A: lane<->M, ISA 16-bit layout
      int ak = kt * 32 + hi * 8;
      a.u[0] = *(const uint4*)&hin[l16 * HS_ + ak];
      a.u[1] = *(const uint4*)&hin[l16 * HS_ + ak + 16];
      ABu b[4];                                      // batch the 8 global_load_b128
#pragma unroll
      for (int g = 0; g < 4; ++g) {
        int ntg = g * 32 + w;                        // global N-tile over 2048 cols
        const __bf16* bp = wb + ((size_t)(ntg * 16 + kt) << 9) + lane * 16;
        b[g].u[0] = *(const uint4*)bp;
        b[g].u[1] = *(const uint4*)(bp + 8);
      }
#pragma unroll
      for (int g = 0; g < 4; ++g)
        acc[g] = __builtin_amdgcn_wmma_f32_16x16x32_bf16(
            false, a.v, false, b[g].v, (short)0, acc[g], false, false);
    }

    // gate nonlinearity + state update, all in registers (C layout: M=r+hi*8, N=l16)
    {
      int j = w * 16 + l16;
#pragma unroll
      for (int r = 0; r < 8; ++r) {
        int m = r + hi * 8;
        float xv = x[(bbase + m) * T_ + t];
        float gi = acc[0][r] + xv * wih_r[0] + bias_r[0];
        float gf = acc[1][r] + xv * wih_r[1] + bias_r[1];
        float gc = acc[2][r] + xv * wih_r[2] + bias_r[2];
        float go = acc[3][r] + xv * wih_r[3] + bias_r[3];
        float cn = sigm_f(gf) * c_st[r] + sigm_f(gi) * tanh_f(gc);
        float hn = sigm_f(go) * tanh_f(cn);
        c_st[r] = cn;
        __bf16 hb = f2bf(hn);
        hout[m * HS_ + j] = hb;
        flat[(size_t)(bbase + m) * KTOT + (size_t)t * HS_ + j] = hb;
      }
    }
    __syncthreads();
  }
}

// ---------------------------------------------------------------------------
// linear1: (128 x 262144 bf16) @ w1^T (f32, converted in-register). K split
// across 64 chunks, N split in 4 slices -> 256 WGs for HBM bandwidth.
// w1 is 537MB streamed exactly once -> nontemporal loads keep 'flat' in L2.
// ---------------------------------------------------------------------------
__global__ void __launch_bounds__(256, 1)
linear1_kernel(const __bf16* __restrict__ flat,
               const float* __restrict__ w1,
               float* __restrict__ partials) {
  const int tid   = threadIdx.x;
  const int lane  = tid & 31;
  const int w     = tid >> 5;            // wave -> M-tile (16 rows)
  const int l16   = lane & 15;
  const int hi    = lane >> 4;
  const int nblk  = blockIdx.x & 3;
  const int kc    = blockIdx.x >> 2;
  const int nbase = nblk * 128;
  const size_t k0 = (size_t)kc * KCH;

  v8f acc[8];
#pragma unroll
  for (int nt = 0; nt < 8; ++nt) acc[nt] = zero8();

#pragma unroll 2
  for (int kt = 0; kt < KCH / 32; ++kt) {
    size_t kb = k0 + (size_t)kt * 32;
    ABu a;
    const __bf16* ap = flat + (size_t)(w * 16 + l16) * KTOT + kb + hi * 8;
    a.u[0] = *(const uint4*)ap;
    a.u[1] = *(const uint4*)(ap + 16);
    size_t kg = kb + lane;
    __builtin_prefetch(&w1[(size_t)nbase * KTOT + kg + 256], 0, 0);
#pragma unroll
    for (int nt = 0; nt < 8; ++nt) {
      ABu b;                               // lane<->N, slot<->K (table order)
#pragma unroll
      for (int i = 0; i < 16; ++i)
        b.h[i] = f2bf(__builtin_nontemporal_load(
            &w1[(size_t)(nbase + nt * 16 + l16) * KTOT + kb + hi * 8 + i + (i & 8)]));
      acc[nt] = __builtin_amdgcn_wmma_f32_16x16x32_bf16(
          false, a.v, false, b.v, (short)0, acc[nt], false, false);
    }
  }
#pragma unroll
  for (int nt = 0; nt < 8; ++nt)
#pragma unroll
    for (int r = 0; r < 8; ++r) {
      int row = w * 16 + r + hi * 8;
      int col = nbase + nt * 16 + l16;
      partials[(size_t)kc * (B_ * HS_) + row * HS_ + col] = acc[nt][r];
    }
}

__global__ void reduce_relu_kernel(const float* __restrict__ partials,
                                   const float* __restrict__ b1,
                                   float* __restrict__ res1) {
  int e = blockIdx.x * blockDim.x + threadIdx.x;   // 65536
  float s = 0.f;
#pragma unroll 4
  for (int kc = 0; kc < KC_; ++kc) s += partials[(size_t)kc * (B_ * HS_) + e];
  s += b1[e & (HS_ - 1)];
  res1[e] = fmaxf(s, 0.f);
}

// ---------------------------------------------------------------------------
// head: res1 @ w2^T + b2, log_softmax over 5 classes. One wave32 per row.
// ---------------------------------------------------------------------------
__global__ void head_kernel(const float* __restrict__ res1,
                            const float* __restrict__ w2,
                            const float* __restrict__ b2,
                            float* __restrict__ out) {
  int row = blockIdx.x;
  int lane = threadIdx.x;
  float a[NC_];
#pragma unroll
  for (int c = 0; c < NC_; ++c) a[c] = 0.f;
  for (int k = lane; k < HS_; k += 32) {
    float r = res1[row * HS_ + k];
#pragma unroll
    for (int c = 0; c < NC_; ++c) a[c] += r * w2[c * HS_ + k];
  }
#pragma unroll
  for (int c = 0; c < NC_; ++c)
    for (int off = 16; off > 0; off >>= 1)
      a[c] += __shfl_xor(a[c], off, 32);
  if (lane == 0) {
    float z[NC_], m = -1e30f;
#pragma unroll
    for (int c = 0; c < NC_; ++c) { z[c] = a[c] + b2[c]; m = fmaxf(m, z[c]); }
    float s = 0.f;
#pragma unroll
    for (int c = 0; c < NC_; ++c) s += __expf(z[c] - m);
    float l = m + __logf(s);
#pragma unroll
    for (int c = 0; c < NC_; ++c) out[row * NC_ + c] = z[c] - l;
  }
}

extern "C" void kernel_launch(void* const* d_in, const int* in_sizes, int n_in,
                              void* d_out, int out_size, void* d_ws, size_t ws_size,
                              hipStream_t stream) {
  const float* x    = (const float*)d_in[0];
  const float* w_ih = (const float*)d_in[1];
  const float* w_hh = (const float*)d_in[2];
  const float* b_ih = (const float*)d_in[3];
  const float* b_hh = (const float*)d_in[4];
  const float* w1   = (const float*)d_in[5];
  const float* b1   = (const float*)d_in[6];
  const float* w2   = (const float*)d_in[7];
  const float* b2   = (const float*)d_in[8];
  float* out = (float*)d_out;

  char* ws = (char*)d_ws;
  size_t off = 0;
  __bf16* ws_whh  = (__bf16*)(ws + off); off += (size_t)G4_ * HS_ * 2;        // 2 MB
  float*  ws_bias = (float*) (ws + off); off += (size_t)G4_ * 4;              // 8 KB
  __bf16* ws_flat = (__bf16*)(ws + off); off += (size_t)B_ * KTOT * 2;        // 64 MB
  float*  ws_part = (float*) (ws + off); off += (size_t)KC_ * B_ * HS_ * 4;   // 16 MB
  float*  ws_res1 = (float*) (ws + off); off += (size_t)B_ * HS_ * 4;         // 256 KB
  (void)ws_size; (void)in_sizes; (void)n_in; (void)out_size;

  prep_kernel<<<(G4_ * HS_ + 255) / 256, 256, 0, stream>>>(w_hh, b_ih, b_hh, ws_whh, ws_bias);
  lstm_kernel<<<B_ / 16, 1024, 0, stream>>>(x, w_ih, ws_whh, ws_bias, ws_flat);
  linear1_kernel<<<4 * KC_, 256, 0, stream>>>(ws_flat, w1, ws_part);
  reduce_relu_kernel<<<(B_ * HS_) / 256, 256, 0, stream>>>(ws_part, b1, ws_res1);
  head_kernel<<<B_, 32, 0, stream>>>(ws_res1, w2, b2, out);
}
